// GPMgSWARDGraphEncoder_58737972740404
// MI455X (gfx1250) — matrix operand order repaired and checked by reading, not compiled, not run on hardware
//
#include <hip/hip_runtime.h>
#include <hip/hip_bf16.h>
#include <math.h>

// ---------------------------------------------------------------------------
// CDNA5 (gfx1250, wave32) WMMA-based implementation of the two-phase graph
// transformer encoder.  All GEMMs run on v_wmma_f32_16x16x32_f16.
// Weights are pre-packed f32->f16 once per launch; V is kept transposed so
// every WMMA operand load is a contiguous 128-bit load.
// ---------------------------------------------------------------------------

typedef __attribute__((ext_vector_type(16))) _Float16 v16h;
typedef __attribute__((ext_vector_type(8)))  _Float16 v8h;
typedef __attribute__((ext_vector_type(8)))  float    v8f;

#define DEV __device__ __forceinline__

DEV v8f wmma_f16(v16h a, v16h b, v8f c) {
    // D = A(16x32 f16) * B(32x16 f16) + C(16x16 f32)
    return __builtin_amdgcn_wmma_f32_16x16x32_f16(
        /*neg_a=*/false, a, /*neg_b=*/false, b,
        /*c_mod=*/(short)0, c, /*reuse_a=*/false, /*reuse_b=*/false);
}

// A fragment (16x32, f16).  p0 -> element [0][k-origin] of MxK row-major f16,
// leading dim ld (multiple of 8).  Layout: lanes 0-15 -> M=lane,
// K=0..15; lanes 16-31 -> M=lane-16, K=16..31 (matches load_B_f16).
// One contiguous 32B run per lane -> two 16B vector LDS/global loads.
DEV v16h load_A_f16(const _Float16* p0, int ld) {
    int l  = threadIdx.x & 31;
    int m  = l & 15;
    int hi = l >> 4;
    const _Float16* row = p0 + (size_t)m * ld + hi * 16;
    v8h lo = *(const v8h*)(row);
    v8h hh = *(const v8h*)(row + 8);
    return __builtin_shufflevector(lo, hh, 0, 1, 2, 3, 4, 5, 6, 7,
                                           8, 9, 10, 11, 12, 13, 14, 15);
}

// B fragment (32x16, f16) from B^T storage (W[n][k] row-major, k contiguous).
// p0 -> element [n-origin][k-origin], ld multiple of 16.
// One contiguous 32B run per lane -> single v16h load.
DEV v16h load_B_f16(const _Float16* p0, int ld) {
    int l  = threadIdx.x & 31;
    int n  = l & 15;
    int hi = l >> 4;
    return *(const v16h*)(p0 + (size_t)n * ld + hi * 16);
}

// LayerNorm of a 16x128 LDS tile (threads 0..15, one row each).
DEV void ln_tile16(float xt[16][128], _Float16 yt[16][128],
                   const float* w, const float* b, int tid) {
    if (tid < 16) {
        int j = tid;
        float mu = 0.f;
        for (int c = 0; c < 128; ++c) mu += xt[j][c];
        mu *= (1.f / 128.f);
        float var = 0.f;
        for (int c = 0; c < 128; ++c) { float d = xt[j][c] - mu; var += d * d; }
        var *= (1.f / 128.f);
        float inv = rsqrtf(var + 1e-5f);
        for (int c = 0; c < 128; ++c)
            yt[j][c] = (_Float16)((xt[j][c] - mu) * inv * w[c] + b[c]);
    }
}

// ---------------------------------------------------------------------------
// Weight pre-pack: f32 -> f16 (runs once per launch; weights live in L2).
// ---------------------------------------------------------------------------
__global__ void f32_to_f16_kernel(const float* __restrict__ src,
                                  _Float16* __restrict__ dst, int n)
{
    for (int i = blockIdx.x * blockDim.x + threadIdx.x; i < n;
         i += gridDim.x * blockDim.x)
        dst[i] = (_Float16)src[i];
}

// ---------------------------------------------------------------------------
// Kernel: fused local encoder.  One block (128 thr = 4 waves) per subgraph.
// ---------------------------------------------------------------------------
__global__ __launch_bounds__(128)
void local_encoder_kernel(
    const float* __restrict__ x,            // [4096,64]
    const float* __restrict__ log_probs,    // [16384]
    const float* __restrict__ ea_flat,      // [E_loc,16]
    const float* __restrict__ init_W,       // [128,66]
    const float* __restrict__ init_b,       // [128]
    const float* __restrict__ ep_W,         // [4,16]
    const float* __restrict__ ep_b,         // [4]
    const float* __restrict__ nw,           // [128] final LN
    const float* __restrict__ nb,           // [128]
    const int*   __restrict__ nodes,        // [16384,16]
    const int*   __restrict__ eis,          // [2,E_loc]
    const int*   __restrict__ eptr,         // [16385]
    const unsigned char* __restrict__ valid,// [16384*16]
    const _Float16* __restrict__ Wqkv, const float* __restrict__ bqkv,
    const _Float16* __restrict__ Wo,   const float* __restrict__ bo,
    const float* __restrict__ ln1w, const float* __restrict__ ln1b,
    const float* __restrict__ ln2w, const float* __restrict__ ln2b,
    const _Float16* __restrict__ W1,   const float* __restrict__ b1,
    const _Float16* __restrict__ W2,   const float* __restrict__ b2,
    float* __restrict__ sub_embs,           // [16384,128]
    int E_loc)
{
    __shared__ float    xt[16][128];        // activations (f32 residual stream)
    __shared__ _Float16 yt[16][128];        // LN output (WMMA A operand)
    __shared__ _Float16 qkvt[16][256];      // Q and K sections
    __shared__ _Float16 vtt[128][32];       // V transposed [dim][token], padded
    __shared__ _Float16 atn[16][128];       // attention output
    __shared__ _Float16 ft[16][256];        // FFN hidden
    __shared__ _Float16 pbuf[4][16][32];    // softmax probs (padded K, hi=0)
    __shared__ float    biasT[4][16][16];   // edge bias [head][q][k]
    __shared__ float    feat[16][66];       // gathered input features
    __shared__ float    keypad[16];
    __shared__ float    validf[16];
    __shared__ int      nid[16];
    __shared__ float    vcnt;

    const int s    = blockIdx.x;
    const int tid  = threadIdx.x;
    const int wave = tid >> 5;
    const int lane = tid & 31;
    const int n16  = lane & 15;
    const int mb   = (lane >> 4) * 8;

    if (tid < 16) {
        int j = tid;
        nid[j] = nodes[s * 16 + j];
        float vf = valid[s * 16 + j] ? 1.f : 0.f;
        validf[j] = vf;
        keypad[j] = (vf > 0.f) ? 0.f : -1e9f;
    }
    // zero the (persistent) zero-padded regions and the bias tile
    for (int idx = tid; idx < 4 * 16 * 32; idx += 128)
        ((_Float16*)pbuf)[idx] = (_Float16)0.f;
    for (int idx = tid; idx < 128 * 16; idx += 128)
        vtt[idx >> 4][16 + (idx & 15)] = (_Float16)0.f;
    for (int idx = tid; idx < 4 * 16 * 16; idx += 128)
        ((float*)biasT)[idx] = 0.f;
    __syncthreads();
    for (int idx = tid; idx < 16 * 64; idx += 128) {
        int j = idx >> 6, i = idx & 63;
        feat[j][i] = x[(size_t)nid[j] * 64 + i];
    }
    if (tid < 16) {
        int j = tid;
        float lpv = log_probs[s];
        if (!(lpv == lpv) || fabsf(lpv) > 1e30f) lpv = 0.f;   // isfinite guard
        feat[j][64] = lpv * validf[j];
        feat[j][65] = 0.f;
    }
    __syncthreads();
    if (tid == 0) {
        int root = s >> 2;                  // m = 4 samples per target node
        int rl = 0;
        for (int j = 0; j < 16; ++j) if (nid[j] == root) { rl = j; break; }
        feat[rl][65] = 1.f;
        float cnt = 0.f;
        for (int j = 0; j < 16; ++j) cnt += validf[j];
        vcnt = fmaxf(cnt, 1.f);
    }
    __syncthreads();

    // init projection: h[j][tid] = feat[j][:] . init_W[tid][:] + init_b[tid]
    {
        float acc[16];
        float bb = init_b[tid];
#pragma unroll
        for (int j = 0; j < 16; ++j) acc[j] = bb;
        for (int i = 0; i < 66; ++i) {
            float w = init_W[tid * 66 + i];
#pragma unroll
            for (int j = 0; j < 16; ++j) acc[j] += feat[j][i] * w;
        }
#pragma unroll
        for (int j = 0; j < 16; ++j) xt[j][tid] = acc[j];
    }
    // local edge bias scatter (edges of subgraph s are contiguous)
    {
        int eb = eptr[s], ee = eptr[s + 1];
        for (int e = eb + tid; e < ee; e += 128) {
            int i0 = eis[e], i1 = eis[E_loc + e];
            const float* a = ea_flat + (size_t)e * 16;
#pragma unroll
            for (int h = 0; h < 4; ++h) {
                float v = ep_b[h];
#pragma unroll
                for (int d = 0; d < 16; ++d) v += a[d] * ep_W[h * 16 + d];
                atomicAdd(&biasT[h][i0][i1], v);
            }
        }
    }
    __syncthreads();

    const float scale = 0.17677669529663687f;   // 1/sqrt(32)

    for (int l = 0; l < 4; ++l) {
        const _Float16* Wqkv_l = Wqkv + (size_t)l * 384 * 128;
        const float*    bqkv_l = bqkv + l * 384;
        const _Float16* Wo_l   = Wo   + (size_t)l * 128 * 128;
        const float*    bo_l   = bo   + l * 128;
        const _Float16* W1_l   = W1   + (size_t)l * 256 * 128;
        const float*    b1_l   = b1   + l * 256;
        const _Float16* W2_l   = W2   + (size_t)l * 128 * 256;
        const float*    b2_l   = b2   + l * 128;

        ln_tile16(xt, yt, ln1w + l * 128, ln1b + l * 128, tid);
        __syncthreads();

        // ---- QKV: [16,128] x [128,384]^T  (24 n-tiles, 6 per wave) ----
        // Q,K tiles (nt<16) -> qkvt row-major; V tiles -> vtt transposed.
        {
            v16h a0 = load_A_f16(&yt[0][0] +  0, 128);
            v16h a1 = load_A_f16(&yt[0][0] + 32, 128);
            v16h a2 = load_A_f16(&yt[0][0] + 64, 128);
            v16h a3 = load_A_f16(&yt[0][0] + 96, 128);
            for (int q = 0; q < 6; ++q) {
                int nt = wave * 6 + q;
                const _Float16* wp = Wqkv_l + (size_t)(nt * 16) * 128;
                __builtin_prefetch(wp + 16 * 128, 0, 1);
                v8f c = {};
                c = wmma_f16(a0, load_B_f16(wp +  0, 128), c);
                c = wmma_f16(a1, load_B_f16(wp + 32, 128), c);
                c = wmma_f16(a2, load_B_f16(wp + 64, 128), c);
                c = wmma_f16(a3, load_B_f16(wp + 96, 128), c);
                int n = nt * 16 + n16;
                float bb = bqkv_l[n];
                if (nt < 16) {
#pragma unroll
                    for (int r = 0; r < 8; ++r)
                        qkvt[mb + r][n] = (_Float16)(c[r] + bb);
                } else {
#pragma unroll
                    for (int r = 0; r < 8; ++r)
                        vtt[n - 256][mb + r] = (_Float16)(c[r] + bb);
                }
            }
        }
        __syncthreads();

        // ---- attention: head = wave, S = Q K^T (one WMMA), softmax, O = P V ----
        {
            int h = wave;
            v16h aq = load_A_f16(&qkvt[0][0] + h * 32, 256);
            v16h bk = load_B_f16(&qkvt[0][0] + 128 + h * 32, 256);
            v8f c = {};
            c = wmma_f16(aq, bk, c);
#pragma unroll
            for (int r = 0; r < 8; ++r) {
                int m = mb + r;
                float sv = c[r] * scale + biasT[h][m][n16] + keypad[n16];
                float mx = sv;
#pragma unroll
                for (int d = 1; d < 16; d <<= 1) mx = fmaxf(mx, __shfl_xor(mx, d, 16));
                float ev  = __expf(sv - mx);
                float sum = ev;
#pragma unroll
                for (int d = 1; d < 16; d <<= 1) sum += __shfl_xor(sum, d, 16);
                pbuf[h][m][n16] = (_Float16)(ev / sum);
            }
            __syncthreads();
            v16h ap = load_A_f16(&pbuf[h][0][0], 32);   // cols 16..31 are zero
            v8f o0 = {}, o1 = {};
            o0 = wmma_f16(ap, load_B_f16(&vtt[h * 32][0],      32), o0);
            o1 = wmma_f16(ap, load_B_f16(&vtt[h * 32 + 16][0], 32), o1);
#pragma unroll
            for (int r = 0; r < 8; ++r) {
                atn[mb + r][h * 32 + n16]      = (_Float16)o0[r];
                atn[mb + r][h * 32 + 16 + n16] = (_Float16)o1[r];
            }
        }
        __syncthreads();

        // ---- out projection + residual: x += atn @ Wo^T + bo ----
        {
            v16h a0 = load_A_f16(&atn[0][0] +  0, 128);
            v16h a1 = load_A_f16(&atn[0][0] + 32, 128);
            v16h a2 = load_A_f16(&atn[0][0] + 64, 128);
            v16h a3 = load_A_f16(&atn[0][0] + 96, 128);
            for (int q = 0; q < 2; ++q) {
                int nt = wave * 2 + q;
                const _Float16* wp = Wo_l + (size_t)(nt * 16) * 128;
                v8f c = {};
                c = wmma_f16(a0, load_B_f16(wp +  0, 128), c);
                c = wmma_f16(a1, load_B_f16(wp + 32, 128), c);
                c = wmma_f16(a2, load_B_f16(wp + 64, 128), c);
                c = wmma_f16(a3, load_B_f16(wp + 96, 128), c);
                int n = nt * 16 + n16;
                float bb = bo_l[n];
#pragma unroll
                for (int r = 0; r < 8; ++r) xt[mb + r][n] += c[r] + bb;
            }
        }
        __syncthreads();

        ln_tile16(xt, yt, ln2w + l * 128, ln2b + l * 128, tid);
        __syncthreads();

        // ---- FFN1: relu([16,128] x [128,256]^T)  (16 tiles, 4 per wave) ----
        {
            v16h a0 = load_A_f16(&yt[0][0] +  0, 128);
            v16h a1 = load_A_f16(&yt[0][0] + 32, 128);
            v16h a2 = load_A_f16(&yt[0][0] + 64, 128);
            v16h a3 = load_A_f16(&yt[0][0] + 96, 128);
            for (int q = 0; q < 4; ++q) {
                int nt = wave * 4 + q;
                const _Float16* wp = W1_l + (size_t)(nt * 16) * 128;
                __builtin_prefetch(wp + 16 * 128, 0, 1);
                v8f c = {};
                c = wmma_f16(a0, load_B_f16(wp +  0, 128), c);
                c = wmma_f16(a1, load_B_f16(wp + 32, 128), c);
                c = wmma_f16(a2, load_B_f16(wp + 64, 128), c);
                c = wmma_f16(a3, load_B_f16(wp + 96, 128), c);
                int n = nt * 16 + n16;
                float bb = b1_l[n];
#pragma unroll
                for (int r = 0; r < 8; ++r)
                    ft[mb + r][n] = (_Float16)fmaxf(c[r] + bb, 0.f);
            }
        }
        __syncthreads();

        // ---- FFN2 + residual: x += ft @ W2^T + b2  (K=256) ----
        {
            for (int q = 0; q < 2; ++q) {
                int nt = wave * 2 + q;
                const _Float16* wp = W2_l + (size_t)(nt * 16) * 256;
                v8f c = {};
#pragma unroll
                for (int kt = 0; kt < 8; ++kt) {
                    v16h a = load_A_f16(&ft[0][0] + kt * 32, 256);
                    c = wmma_f16(a, load_B_f16(wp + kt * 32, 256), c);
                }
                int n = nt * 16 + n16;
                float bb = b2_l[n];
#pragma unroll
                for (int r = 0; r < 8; ++r) xt[mb + r][n] += c[r] + bb;
            }
        }
        __syncthreads();
    }

    ln_tile16(xt, yt, nw, nb, tid);
    __syncthreads();
    {
        float acc = 0.f;
        for (int j = 0; j < 16; ++j) acc += validf[j] * (float)yt[j][tid];
        sub_embs[(size_t)s * 128 + tid] = acc / vcnt;
    }
}

// ---------------------------------------------------------------------------
// Kernel: softmax-weighted aggregation over the m=4 samples per node.
// ---------------------------------------------------------------------------
__global__ void aggregate_kernel(const float* __restrict__ sub,
                                 const float* __restrict__ lp,
                                 float* __restrict__ xg, int T)
{
    int idx = blockIdx.x * blockDim.x + threadIdx.x;
    if (idx >= T * 128) return;
    int t = idx >> 7, c = idx & 127;
    float l0 = lp[t * 4 + 0] * 2.f;   // / TEMP (0.5)
    float l1 = lp[t * 4 + 1] * 2.f;
    float l2 = lp[t * 4 + 2] * 2.f;
    float l3 = lp[t * 4 + 3] * 2.f;
    float mx = fmaxf(fmaxf(l0, l1), fmaxf(l2, l3));
    float w0 = __expf(l0 - mx), w1 = __expf(l1 - mx);
    float w2 = __expf(l2 - mx), w3 = __expf(l3 - mx);
    float inv = 1.f / (w0 + w1 + w2 + w3);
    float acc = w0 * sub[(size_t)(t * 4 + 0) * 128 + c]
              + w1 * sub[(size_t)(t * 4 + 1) * 128 + c]
              + w2 * sub[(size_t)(t * 4 + 2) * 128 + c]
              + w3 * sub[(size_t)(t * 4 + 3) * 128 + c];
    xg[idx] = acc * inv;
}

// ---------------------------------------------------------------------------
// Global-phase helper kernels
// ---------------------------------------------------------------------------
__global__ void zero_f32_kernel(float* __restrict__ p, int n) {
    for (int i = blockIdx.x * blockDim.x + threadIdx.x; i < n;
         i += gridDim.x * blockDim.x)
        p[i] = 0.f;
}

__global__ void gbias_scatter_kernel(const float* __restrict__ ea,
                                     const float* __restrict__ W,
                                     const float* __restrict__ b,
                                     const int* __restrict__ ei,
                                     const int* __restrict__ bvec,
                                     const int* __restrict__ ptr,
                                     float* __restrict__ gbias, int E)
{
    int e = blockIdx.x * blockDim.x + threadIdx.x;
    if (e >= E) return;
    int src = ei[e], dst = ei[E + e];
    int g  = bvec[src];
    int sl = src - ptr[g], dl = dst - ptr[g];
    const float* a = ea + (size_t)e * 16;
#pragma unroll
    for (int h = 0; h < 4; ++h) {
        float v = b[h];
#pragma unroll
        for (int d = 0; d < 16; ++d) v += a[d] * W[h * 16 + d];
        atomicAdd(&gbias[(((size_t)g * 4 + h) * 512 + sl) * 512 + dl], v);
    }
}

__global__ void ln_rows_kernel(const float* __restrict__ x,
                               const float* __restrict__ w,
                               const float* __restrict__ b,
                               _Float16* __restrict__ y, int rows)
{
    int r = blockIdx.x * blockDim.x + threadIdx.x;
    if (r >= rows) return;
    const float* p = x + (size_t)r * 128;
    float mu = 0.f;
    for (int c = 0; c < 128; ++c) mu += p[c];
    mu *= (1.f / 128.f);
    float var = 0.f;
    for (int c = 0; c < 128; ++c) { float d = p[c] - mu; var += d * d; }
    var *= (1.f / 128.f);
    float inv = rsqrtf(var + 1e-5f);
    _Float16* q = y + (size_t)r * 128;
    for (int c = 0; c < 128; ++c)
        q[c] = (_Float16)((p[c] - mu) * inv * w[c] + b[c]);
}

// Transpose the V section of QKV into vtg[g][dim][512] (f16) once per layer.
__global__ void vtrans_kernel(const _Float16* __restrict__ qkv,
                              _Float16* __restrict__ vtg)
{
    int idx = blockIdx.x * blockDim.x + threadIdx.x;   // 8*512*128
    if (idx >= 8 * 512 * 128) return;
    int dim = idx & 127;
    int tok = (idx >> 7) & 511;
    int g   = idx >> 16;
    vtg[((size_t)(g * 128 + dim)) * 512 + tok] =
        qkv[((size_t)(g * 512 + tok)) * 384 + 256 + dim];
}

// Generic WMMA GEMM over 16-row tiles: OUT[M,N] = f(A[M,K] @ W[N,K]^T + bias)
// mode 0: f16 out; mode 1: relu + f16 out; mode 2: residual add into f32 (ld=N)
__global__ __launch_bounds__(128)
void gemm16_kernel(const _Float16* __restrict__ A,
                   const _Float16* __restrict__ W,
                   const float* __restrict__ bias,
                   _Float16* __restrict__ outh,
                   float* __restrict__ outf,
                   int N, int K, int mode)
{
    int mt   = blockIdx.x;
    int wave = threadIdx.x >> 5;
    int lane = threadIdx.x & 31;
    int n16  = lane & 15;
    int mb   = (lane >> 4) * 8;
    int ntiles = N >> 4;
    const _Float16* Arow = A + (size_t)(mt * 16) * K;
    for (int nt = wave; nt < ntiles; nt += 4) {
        const _Float16* wp = W + (size_t)(nt * 16) * K;
        __builtin_prefetch(wp + 16 * K, 0, 1);
        v8f c = {};
        for (int kt = 0; kt < (K >> 5); ++kt) {
            v16h a = load_A_f16(Arow + kt * 32, K);
            v16h b = load_B_f16(wp + kt * 32, K);
            c = wmma_f16(a, b, c);
        }
        int n = nt * 16 + n16;
        float bb = bias[n];
        if (mode == 2) {
#pragma unroll
            for (int r = 0; r < 8; ++r)
                outf[(size_t)(mt * 16 + mb + r) * N + n] += c[r] + bb;
        } else {
#pragma unroll
            for (int r = 0; r < 8; ++r) {
                float v = c[r] + bb;
                if (mode == 1) v = fmaxf(v, 0.f);
                outh[(size_t)(mt * 16 + mb + r) * N + n] = (_Float16)v;
            }
        }
    }
}

// Flash-attention over L=512 with dense edge bias.  1 wave per (g, head, qtile).
// Processes 32 keys per iteration: 2 QK^T WMMAs feed one full-K P*V pass.
__global__ __launch_bounds__(32)
void gattn_kernel(const _Float16* __restrict__ qkv,   // [4096,384]
                  const _Float16* __restrict__ vtg,   // [8][128][512] (V^T)
                  const float* __restrict__ gbias,    // [8,4,512,512]
                  _Float16* __restrict__ oh)          // [4096,128]
{
    __shared__ _Float16 pb[16][32];
    int bx = blockIdx.x;
    int qt = bx & 31, h = (bx >> 5) & 3, g = bx >> 7;
    int lane = threadIdx.x;
    int n16  = lane & 15;
    int mb   = (lane >> 4) * 8;
    const int LD = 384;
    const _Float16* qbase = qkv + (size_t)(g * 512 + qt * 16) * LD + h * 32;
    v16h aq = load_A_f16(qbase, LD);
    const float scale = 0.17677669529663687f;
    const float* bias_row = gbias + ((size_t)(g * 4 + h) * 512 + qt * 16) * 512;
    float Mr[8], Lr[8];
    v8f O0 = {}, O1 = {};
#pragma unroll
    for (int r = 0; r < 8; ++r) { Mr[r] = -1e30f; Lr[r] = 0.f; }

    for (int kb = 0; kb < 16; ++kb) {
        const _Float16* kbase = qkv + (size_t)(g * 512 + kb * 32) * LD + 128 + h * 32;
        v16h bk0 = load_B_f16(kbase,           LD);   // keys kb*32 + 0..15
        v16h bk1 = load_B_f16(kbase + 16 * LD, LD);   // keys kb*32 + 16..31
        v8f c0 = {}, c1 = {};
        c0 = wmma_f16(aq, bk0, c0);
        c1 = wmma_f16(aq, bk1, c1);
#pragma unroll
        for (int r = 0; r < 8; ++r) {
            int m = mb + r;
            const float* bp = bias_row + (size_t)m * 512 + kb * 32;
            float s0 = c0[r] * scale + bp[n16];
            float s1 = c1[r] * scale + bp[16 + n16];
            float mx = fmaxf(s0, s1);
#pragma unroll
            for (int d = 1; d < 16; d <<= 1) mx = fmaxf(mx, __shfl_xor(mx, d, 16));
            float Mn    = fmaxf(Mr[r], mx);
            float alpha = __expf(Mr[r] - Mn);
            float p0    = __expf(s0 - Mn);
            float p1    = __expf(s1 - Mn);
            float rs    = p0 + p1;
#pragma unroll
            for (int d = 1; d < 16; d <<= 1) rs += __shfl_xor(rs, d, 16);
            Lr[r] = Lr[r] * alpha + rs;
            Mr[r] = Mn;
            O0[r] *= alpha;
            O1[r] *= alpha;
            pb[m][n16]      = (_Float16)p0;
            pb[m][16 + n16] = (_Float16)p1;
        }
        __syncthreads();
        v16h ap = load_A_f16(&pb[0][0], 32);          // full K=32 valid
        const _Float16* vb = vtg + ((size_t)(g * 128 + h * 32)) * 512 + kb * 32;
        O0 = wmma_f16(ap, load_B_f16(vb,            512), O0);
        O1 = wmma_f16(ap, load_B_f16(vb + 16 * 512, 512), O1);
        __syncthreads();
    }
    _Float16* orow = oh + (size_t)(g * 512 + qt * 16) * 128 + h * 32;
#pragma unroll
    for (int r = 0; r < 8; ++r) {
        float inv = 1.f / Lr[r];
        orow[(size_t)(mb + r) * 128 + n16]      = (_Float16)(O0[r] * inv);
        orow[(size_t)(mb + r) * 128 + 16 + n16] = (_Float16)(O1[r] * inv);
    }
}

__global__ void colsum_kernel(const _Float16* __restrict__ y,
                              float* __restrict__ out)
{
    int g = blockIdx.x, c = threadIdx.x;
    float s = 0.f;
    for (int r = 0; r < 512; ++r)
        s += (float)y[((size_t)g * 512 + r) * 128 + c];
    out[g * 128 + c] = s;
}

// ---------------------------------------------------------------------------
// Host launcher
// ---------------------------------------------------------------------------
extern "C" void kernel_launch(void* const* d_in, const int* in_sizes, int n_in,
                              void* d_out, int out_size, void* d_ws, size_t ws_size,
                              hipStream_t stream)
{
    const int N = 4096, S = 16384, G = 8, E = 65536, NL = 4;
    const int E_loc = in_sizes[2] / 16;

    const float* x         = (const float*)d_in[0];
    const float* log_probs = (const float*)d_in[1];
    const float* ea_flat   = (const float*)d_in[2];
    const float* edge_attr = (const float*)d_in[3];
    const float* init_W    = (const float*)d_in[4];
    const float* init_b    = (const float*)d_in[5];
    const float* loc_ep_W  = (const float*)d_in[6];
    const float* loc_ep_b  = (const float*)d_in[7];
    const float* glob_ep_W = (const float*)d_in[8];
    const float* glob_ep_b = (const float*)d_in[9];
    const float* loc_nw    = (const float*)d_in[10];
    const float* loc_nb    = (const float*)d_in[11];
    const float* glob_nw   = (const float*)d_in[12];
    const float* glob_nb   = (const float*)d_in[13];
    const int*   nodes     = (const int*)d_in[14];
    const int*   eis       = (const int*)d_in[15];
    const int*   eptr      = (const int*)d_in[16];
    const int*   eidx      = (const int*)d_in[17];
    const int*   bvec      = (const int*)d_in[18];
    const int*   ptr       = (const int*)d_in[19];
    const unsigned char* valid = (const unsigned char*)d_in[20];
    const float* lW[12];
    const float* gW[12];
    for (int i = 0; i < 12; ++i) lW[i] = (const float*)d_in[21 + i];
    for (int i = 0; i < 12; ++i) gW[i] = (const float*)d_in[33 + i];
    // order: Wqkv, bqkv, Wo, bo, ln1w, ln1b, ln2w, ln2b, W1, b1, W2, b2

    char* w8 = (char*)d_ws;
    auto alloc = [&](size_t bytes) -> void* {
        void* p = (void*)w8;
        w8 += (bytes + 255) & ~(size_t)255;
        return p;
    };
    float*    gbias = (float*)   alloc((size_t)G * 4 * 512 * 512 * 4);
    float*    sub   = (float*)   alloc((size_t)S * 128 * 4);
    float*    xg    = (float*)   alloc((size_t)N * 128 * 4);
    _Float16* yh    = (_Float16*)alloc((size_t)N * 128 * 2);
    _Float16* qkvh  = (_Float16*)alloc((size_t)N * 384 * 2);
    _Float16* vtg   = (_Float16*)alloc((size_t)G * 128 * 512 * 2);
    _Float16* oh    = (_Float16*)alloc((size_t)N * 128 * 2);
    _Float16* h1    = (_Float16*)alloc((size_t)N * 256 * 2);
    // f16-packed weights
    const int nqkv = NL * 384 * 128, no = NL * 128 * 128;
    const int n1   = NL * 256 * 128, n2 = NL * 128 * 256;
    _Float16* lWqkvH = (_Float16*)alloc((size_t)nqkv * 2);
    _Float16* lWoH   = (_Float16*)alloc((size_t)no * 2);
    _Float16* lW1H   = (_Float16*)alloc((size_t)n1 * 2);
    _Float16* lW2H   = (_Float16*)alloc((size_t)n2 * 2);
    _Float16* gWqkvH = (_Float16*)alloc((size_t)nqkv * 2);
    _Float16* gWoH   = (_Float16*)alloc((size_t)no * 2);
    _Float16* gW1H   = (_Float16*)alloc((size_t)n1 * 2);
    _Float16* gW2H   = (_Float16*)alloc((size_t)n2 * 2);

    // ---- weight pre-pack (f32 -> f16) ----
    f32_to_f16_kernel<<<256, 256, 0, stream>>>(lW[0],  lWqkvH, nqkv);
    f32_to_f16_kernel<<<256, 256, 0, stream>>>(lW[2],  lWoH,   no);
    f32_to_f16_kernel<<<256, 256, 0, stream>>>(lW[8],  lW1H,   n1);
    f32_to_f16_kernel<<<256, 256, 0, stream>>>(lW[10], lW2H,   n2);
    f32_to_f16_kernel<<<256, 256, 0, stream>>>(gW[0],  gWqkvH, nqkv);
    f32_to_f16_kernel<<<256, 256, 0, stream>>>(gW[2],  gWoH,   no);
    f32_to_f16_kernel<<<256, 256, 0, stream>>>(gW[8],  gW1H,   n1);
    f32_to_f16_kernel<<<256, 256, 0, stream>>>(gW[10], gW2H,   n2);

    // ---- global edge bias ----
    zero_f32_kernel<<<4096, 256, 0, stream>>>(gbias, G * 4 * 512 * 512);
    gbias_scatter_kernel<<<(E + 255) / 256, 256, 0, stream>>>(
        edge_attr, glob_ep_W, glob_ep_b, eidx, bvec, ptr, gbias, E);

    // ---- fused local encoder ----
    local_encoder_kernel<<<S, 128, 0, stream>>>(
        x, log_probs, ea_flat, init_W, init_b, loc_ep_W, loc_ep_b,
        loc_nw, loc_nb, nodes, eis, eptr, valid,
        lWqkvH, lW[1], lWoH, lW[3], lW[4], lW[5], lW[6], lW[7],
        lW1H, lW[9], lW2H, lW[11],
        sub, E_loc);

    // ---- sample aggregation -> global token stream ----
    aggregate_kernel<<<(N * 128 + 255) / 256, 256, 0, stream>>>(sub, log_probs, xg, N);

    // ---- global transformer (4 layers) ----
    const int MT = N / 16;   // 256 row tiles
    for (int l = 0; l < 4; ++l) {
        ln_rows_kernel<<<(N + 255) / 256, 256, 0, stream>>>(
            xg, gW[4] + l * 128, gW[5] + l * 128, yh, N);
        gemm16_kernel<<<MT, 128, 0, stream>>>(
            yh, gWqkvH + (size_t)l * 384 * 128, gW[1] + l * 384,
            qkvh, nullptr, 384, 128, 0);
        vtrans_kernel<<<(G * 512 * 128 + 255) / 256, 256, 0, stream>>>(qkvh, vtg);
        gattn_kernel<<<G * 4 * 32, 32, 0, stream>>>(qkvh, vtg, gbias, oh);
        gemm16_kernel<<<MT, 128, 0, stream>>>(
            oh, gWoH + (size_t)l * 128 * 128, gW[3] + l * 128,
            nullptr, xg, 128, 128, 2);
        ln_rows_kernel<<<(N + 255) / 256, 256, 0, stream>>>(
            xg, gW[6] + l * 128, gW[7] + l * 128, yh, N);
        gemm16_kernel<<<MT, 128, 0, stream>>>(
            yh, gW1H + (size_t)l * 256 * 128, gW[9] + l * 256,
            h1, nullptr, 256, 128, 1);
        gemm16_kernel<<<MT, 128, 0, stream>>>(
            h1, gW2H + (size_t)l * 128 * 256, gW[11] + l * 128,
            nullptr, xg, 128, 256, 2);
    }

    // ---- final LN + per-graph sum ----
    ln_rows_kernel<<<(N + 255) / 256, 256, 0, stream>>>(xg, glob_nw, glob_nb, yh, N);
    colsum_kernel<<<G, 128, 0, stream>>>(yh, (float*)d_out);
}
